// UpdateLayer2DU_29145648071072
// MI455X (gfx1250) — matrix operand order repaired and checked by reading, not compiled, not run on hardware
//
#include <hip/hip_runtime.h>
#include <hip/hip_bf16.h>
#include <math.h>

#define USE_ASYNC_LDS 1

// ---------------- model constants (from reference) ----------------
constexpr int Bc  = 2;
constexpr int LUc = 512;
constexpr int LDc = 256;
constexpr int LTc = 1024;
constexpr int Dc  = 256;
constexpr int NHc = 8;
constexpr int HDc = 32;
constexpr int Hc  = Dc * NHc * 3;               // 6144
constexpr float SCALEc = 0.17677669529663687f;  // 1/sqrt(32)

typedef __attribute__((ext_vector_type(16))) _Float16 v16h;
typedef __attribute__((ext_vector_type(8)))  float    v8f;
typedef _Float16 h16;

// ---------------- device helpers ----------------
// async copy of 16 bytes global -> LDS (CDNA5 GLOBAL_LOAD_ASYNC_TO_LDS_B128,
// tracked with ASYNCcnt). Fallback: plain load+ds_store.
__device__ inline void cp16(unsigned* ldsDst, const unsigned* gSrc) {
#if USE_ASYNC_LDS
  unsigned loff = (unsigned)(uintptr_t)(void*)ldsDst;        // LDS byte offset
  unsigned long long ga = (unsigned long long)(uintptr_t)gSrc;
  asm volatile("global_load_async_to_lds_b128 %0, %1, off"
               :: "v"(loff), "v"(ga) : "memory");
#else
  *(uint4*)ldsDst = *(const uint4*)gSrc;
#endif
}
__device__ inline void cp_wait() {
#if USE_ASYNC_LDS
#if __has_builtin(__builtin_amdgcn_s_wait_asynccnt)
  __builtin_amdgcn_s_wait_asynccnt(0);
#else
  asm volatile("s_wait_asynccnt 0" ::: "memory");
#endif
#endif
}

// A fragment (16x32 f16). ISA 7.12.2: lane L: m=L&15, half=L>>4;
// VGPR v holds k-pair kp = (v>>2)*8 + half*4 + (v&3). LDS row stride 20 u32.
__device__ inline v16h frag_a(const unsigned* lds, int mbase, int lane) {
  int m = mbase + (lane & 15);
  int half = lane >> 4;
  union { unsigned u[8]; v16h h; } r;
#pragma unroll
  for (int v = 0; v < 8; ++v) {
    int kp = ((v >> 2) << 3) + (half << 2) + (v & 3);
    r.u[v] = lds[m * 20 + kp];
  }
  return r.h;
}
// B fragment from [N,K] tile: lane L: n=L&15, half=L>>4;
// VGPR v -> kp = (v>>2)*8 + half*4 + (v&3) (same k order as frag_a).
__device__ inline v16h frag_b(const unsigned* lds, int nbase, int lane) {
  int n = nbase + (lane & 15);
  int half = lane >> 4;
  union { unsigned u[8]; v16h h; } r;
#pragma unroll
  for (int v = 0; v < 8; ++v) r.u[v] = lds[n * 20 + ((v >> 2) << 3) + (half << 2) + (v & 3)];
  return r.h;
}

__device__ inline v8f wmma32(v16h a, v16h b, v8f c) {
  return __builtin_amdgcn_wmma_f32_16x16x32_f16(false, a, false, b, (short)0, c,
                                                false, false);
}

__device__ inline float wred_max(float v) {
#pragma unroll
  for (int o = 16; o; o >>= 1) v = fmaxf(v, __shfl_xor(v, o, 32));
  return v;
}
__device__ inline float wred_sum(float v) {
#pragma unroll
  for (int o = 16; o; o >>= 1) v += __shfl_xor(v, o, 32);
  return v;
}

// ---------------- generic WMMA GEMM (f16 in, f16/f32 out) ----------------
// C[z] = act( A[z] @ B[z]^T + bias ) + resid
// A: f16 [M,K] lda; B: f16 [N,K] ldb (always "transposed" layout).
// per-z offsets: off = (z/zdiv)*sX + (z%zdiv)*sX2   (element units)
struct GemmP {
  const h16* A; const h16* Bm; void* C;
  const float* bias; const float* resid;
  long long lda, ldb, ldc, ldr;
  long long sA, sA2, sB, sB2, sC, sC2, sR;
  int K, zdiv, relu, outHalf, storeT;
};

template <int BM, int BN>
__global__ __launch_bounds__((BM / 32) * (BN / 32) * 32)
void gemm_wmma(GemmP p) {
  constexpr int NW = (BM / 32) * (BN / 32);
  constexpr int T  = NW * 32;
  constexpr int WCOLS = BN / 32;
  constexpr int ASZ = BM * 20;     // u32 per A tile (stride 20 keeps 16B align)
  constexpr int BSZ = BN * 20;
  __shared__ __align__(16) unsigned lds[2 * (ASZ + BSZ)];

  const int tid  = threadIdx.x;
  const int lane = tid & 31;
  const int w    = tid >> 5;
  const int wm   = (w / WCOLS) * 32;
  const int wn   = (w % WCOLS) * 32;

  const int z = blockIdx.z;
  const long long zo = z / p.zdiv, zi = z % p.zdiv;
  const h16* A  = p.A  + zo * p.sA + zi * p.sA2;
  const h16* Bm = p.Bm + zo * p.sB + zi * p.sB2;
  char* Cb = (char*)p.C + (zo * p.sC + zi * p.sC2) * (p.outHalf ? 2 : 4);
  const float* R = p.resid ? (p.resid + zo * p.sR) : nullptr;

  const long long bm = (long long)blockIdx.y * BM;
  const long long bn = (long long)blockIdx.x * BN;

  const unsigned* A32 = (const unsigned*)A;   // f16 pairs as u32
  const unsigned* B32 = (const unsigned*)Bm;

  auto stage = [&](int t, unsigned* bufA, unsigned* bufB) {
    long long k0 = (long long)t * 32;
    for (int i = tid; i < BM * 4; i += T) {        // 4 x b128 per A row
      int m = i >> 2, q = i & 3;
      cp16(bufA + m * 20 + q * 4,
           A32 + (((bm + m) * p.lda + k0) >> 1) + q * 4);
    }
    for (int i = tid; i < BN * 4; i += T) {        // 4 x b128 per B row
      int n = i >> 2, q = i & 3;
      cp16(bufB + n * 20 + q * 4,
           B32 + (((bn + n) * p.ldb + k0) >> 1) + q * 4);
    }
  };

  v8f acc[2][2];
#pragma unroll
  for (int i = 0; i < 2; ++i)
#pragma unroll
    for (int j = 0; j < 2; ++j) acc[i][j] = (v8f)(0.0f);

  const int nsteps = p.K >> 5;
  stage(0, lds, lds + ASZ);
  for (int t = 0; t < nsteps; ++t) {
    cp_wait();                      // own async copies for tile t complete
    __syncthreads();                // everyone's tile t visible
    if (t + 1 < nsteps) {
      unsigned* nb = lds + ((t + 1) & 1) * (ASZ + BSZ);
      stage(t + 1, nb, nb + ASZ);   // overlap with compute of tile t
    }
    const unsigned* bA = lds + (t & 1) * (ASZ + BSZ);
    const unsigned* bB = bA + ASZ;

    v16h a0 = frag_a(bA, wm, lane);
    v16h a1 = frag_a(bA, wm + 16, lane);
    v16h b0 = frag_b(bB, wn, lane);
    v16h b1 = frag_b(bB, wn + 16, lane);
    acc[0][0] = wmma32(a0, b0, acc[0][0]);
    acc[0][1] = wmma32(a0, b1, acc[0][1]);
    acc[1][0] = wmma32(a1, b0, acc[1][0]);
    acc[1][1] = wmma32(a1, b1, acc[1][1]);
  }

#pragma unroll
  for (int mi = 0; mi < 2; ++mi)
#pragma unroll
    for (int ni = 0; ni < 2; ++ni) {
      long long gn = bn + wn + ni * 16 + (lane & 15);
      float bv = p.bias ? p.bias[gn] : 0.0f;
#pragma unroll
      for (int r = 0; r < 8; ++r) {
        long long gm = bm + wm + mi * 16 + r + 8 * (lane >> 4);
        float v = acc[mi][ni][r] + bv;
        if (p.relu) v = fmaxf(v, 0.0f);
        if (R) v += R[gm * p.ldr + gn];
        long long idx = p.storeT ? (gn * p.ldc + gm) : (gm * p.ldc + gn);
        if (p.outHalf) ((h16*)Cb)[idx] = (h16)v;
        else           ((float*)Cb)[idx] = v;
      }
    }
}

// ---------------- conversion kernels ----------------
__global__ void cvt_kernel(const float* s, h16* d, long long n) {
  long long i = (long long)blockIdx.x * blockDim.x + threadIdx.x;
  if (i < n) d[i] = (h16)s[i];
}
// f32 [K,N] -> f16 [N,K] (transpose for the GEMM's B layout)
__global__ void cvtT_kernel(const float* s, h16* d, int K, int N) {
  long long i = (long long)blockIdx.x * blockDim.x + threadIdx.x;
  if (i >= (long long)K * N) return;
  int n = (int)(i / K), k = (int)(i % K);
  d[i] = (h16)s[(long long)k * N + n];
}

// ---------------- small model kernels ----------------
__global__ void rowvec_kernel(const float* wrow, const float* Wk,
                              const float* Wv, float* wk, float* wv) {
  int n = blockIdx.x * blockDim.x + threadIdx.x;
  if (n >= Dc) return;
  float a = 0.f, b = 0.f;
  for (int k = 0; k < Dc; ++k) {
    float x = wrow[k];
    a += x * Wk[k * Dc + n];
    b += x * Wv[k * Dc + n];
  }
  wk[n] = a; wv[n] = b;
}

__global__ void qw_kernel(const h16* Qu, const float* wk, float* qw) {
  int idx = blockIdx.x * blockDim.x + threadIdx.x;
  if (idx >= Bc * LDc * NHc) return;
  int h = idx % NHc, bl = idx / NHc;
  const h16* q = Qu + (long long)bl * Dc + h * HDc;
  const float* w = wk + h * HDc;
  float s = 0.f;
#pragma unroll
  for (int d = 0; d < HDc; ++d) s += (float)q[d] * w[d];
  qw[idx] = s;
}

// uu softmax with rank-1 CI correction; writes f16 att + sOut = sum(att*CI).
__global__ void softmax_ci_kernel(float* S, const float* CI, const float* qw,
                                  h16* att, float* sOut) {
  int row = blockIdx.x;            // (b*NH + h)*LD + l
  int lane = threadIdx.x;
  int l = row % LDc;
  int h = (row / LDc) % NHc;
  int b = row / (LDc * NHc);
  float* srow = S + (long long)row * LUc;
  h16* arow = att + (long long)row * LUc;
  const float* cic = CI + (long long)b * LUc * LDc + l;   // stride LD over u
  float q = qw[(b * LDc + l) * NHc + h];

  float mx = -INFINITY;
  for (int u = lane; u < LUc; u += 32) {
    float v = (srow[u] + cic[(long long)u * LDc] * q) * SCALEc;
    srow[u] = v;
    mx = fmaxf(mx, v);
  }
  mx = wred_max(mx);
  float sum = 0.f;
  for (int u = lane; u < LUc; u += 32) {
    float e = __expf(srow[u] - mx);
    srow[u] = e;
    sum += e;
  }
  sum = wred_sum(sum);
  float inv = 1.0f / sum;
  float sc = 0.f;
  for (int u = lane; u < LUc; u += 32) {
    float a = srow[u] * inv;
    arow[u] = (h16)a;
    sc += a * cic[(long long)u * LDc];
  }
  sc = wred_sum(sc);
  if (lane == 0) sOut[row] = sc;
}

__global__ void softmax_rows_kernel(float* S, h16* att, int n, float scale) {
  int row = blockIdx.x;
  int lane = threadIdx.x;
  float* srow = S + (long long)row * n;
  h16* arow = att + (long long)row * n;
  float mx = -INFINITY;
  for (int u = lane; u < n; u += 32) {
    float v = srow[u] * scale;
    srow[u] = v;
    mx = fmaxf(mx, v);
  }
  mx = wred_max(mx);
  float sum = 0.f;
  for (int u = lane; u < n; u += 32) {
    float e = __expf(srow[u] - mx);
    srow[u] = e;
    sum += e;
  }
  sum = wred_sum(sum);
  float inv = 1.0f / sum;
  for (int u = lane; u < n; u += 32) arow[u] = (h16)(srow[u] * inv);
}

// O[b,l,h*32+d] += s[b,h,l] * wv_row[h*32+d]   (f16 in-place)
__global__ void fixo_kernel(h16* O, const float* s, const float* wv) {
  int idx = blockIdx.x * blockDim.x + threadIdx.x;
  if (idx >= Bc * LDc * Dc) return;
  int n = idx % Dc, bl = idx / Dc;
  int b = bl / LDc, l = bl % LDc;
  int h = n / HDc;
  O[idx] = (h16)((float)O[idx] + s[(b * NHc + h) * LDc + l] * wv[n]);
}

// ---------------- host ----------------
extern "C" void kernel_launch(void* const* d_in, const int* in_sizes, int n_in,
                              void* d_out, int out_size, void* d_ws,
                              size_t ws_size, hipStream_t stream) {
  (void)in_sizes; (void)n_in; (void)out_size; (void)ws_size;
  const float* UU   = (const float*)d_in[0];
  const float* DU   = (const float*)d_in[1];
  const float* TA   = (const float*)d_in[2];
  const float* CI   = (const float*)d_in[3];
  const float* W[12];
  for (int i = 0; i < 12; ++i) W[i] = (const float*)d_in[4 + i];
  const float* ciW  = (const float*)d_in[16];
  const float* cib  = (const float*)d_in[17];
  const float* woW  = (const float*)d_in[18];
  const float* wob  = (const float*)d_in[19];
  const float* rnW1 = (const float*)d_in[20];
  const float* rnb1 = (const float*)d_in[21];
  const float* rnW2 = (const float*)d_in[22];
  const float* rnb2 = (const float*)d_in[23];
  const float* poW  = (const float*)d_in[24];
  const float* pob  = (const float*)d_in[25];
  float* out = (float*)d_out;

  // byte bump allocator, 256B aligned
  char* wsb = (char*)d_ws;
  size_t off = 0;
  auto alloc = [&](size_t bytes) {
    void* p = wsb + off;
    off = (off + bytes + 255) & ~(size_t)255;
    return p;
  };
  auto allocH = [&](long long n) { return (h16*)alloc((size_t)n * 2); };
  auto allocF = [&](long long n) { return (float*)alloc((size_t)n * 4); };

  h16* WT[12];
  for (int i = 0; i < 12; ++i) WT[i] = allocH((long long)Dc * Dc);
  h16* ciWT   = allocH((long long)Dc * Dc);
  h16* woWT   = allocH((long long)Dc * 3 * Dc);
  h16* rnW1T  = allocH((long long)Hc * Dc);
  h16* rnW2T  = allocH((long long)Dc * Hc);
  h16* poWT   = allocH((long long)Dc * Dc);
  h16* UU16   = allocH((long long)Bc * LUc * Dc);
  h16* DU16   = allocH((long long)Bc * LDc * Dc);
  h16* TA16   = allocH((long long)Bc * LTc * Dc);
  h16* base16 = allocH((long long)Bc * LUc * Dc);
  h16* Qb16   = allocH((long long)Bc * LDc * Dc);
  h16* Kb16   = allocH((long long)Bc * LTc * Dc);   // arena (max LT)
  h16* VbT    = allocH((long long)Bc * Dc * LTc);   // V^T arena
  float* S    = allocF((long long)Bc * NHc * LDc * LTc);
  h16* att16  = allocH((long long)Bc * NHc * LDc * LTc);
  h16* O16    = allocH((long long)Bc * LDc * Dc);
  h16* cat16  = allocH((long long)Bc * LDc * 3 * Dc);
  h16* cato16 = allocH((long long)Bc * LDc * Dc);
  h16* h116   = allocH((long long)Bc * LDc * Hc);
  h16* rn16   = allocH((long long)Bc * LDc * Dc);
  float* wkr  = allocF(Dc);
  float* wvr  = allocF(Dc);
  float* qw   = allocF((long long)Bc * LDc * NHc);
  float* sred = allocF((long long)Bc * NHc * LDc);

  auto cvt = [&](const float* s, h16* d, long long n) {
    cvt_kernel<<<(unsigned)((n + 255) / 256), 256, 0, stream>>>(s, d, n);
  };
  auto cvtT = [&](const float* s, h16* d, int K, int N) {
    long long n = (long long)K * N;
    cvtT_kernel<<<(unsigned)((n + 255) / 256), 256, 0, stream>>>(s, d, K, N);
  };

  // -------- conversions --------
  cvt(UU, UU16, (long long)Bc * LUc * Dc);
  cvt(DU, DU16, (long long)Bc * LDc * Dc);
  cvt(TA, TA16, (long long)Bc * LTc * Dc);
  for (int i = 0; i < 12; ++i) cvtT(W[i], WT[i], Dc, Dc);
  cvtT(ciW, ciWT, Dc, Dc);          // first D rows of ci_W
  cvtT(woW, woWT, 3 * Dc, Dc);
  cvtT(rnW1, rnW1T, Dc, Hc);
  cvtT(rnW2, rnW2T, Hc, Dc);
  cvtT(poW, poWT, Dc, Dc);

  auto gemm = [&](int narrow, const h16* A, long long lda, long long sA,
                  long long sA2, const h16* Bm, long long ldb, long long sB,
                  long long sB2, void* C, long long ldc, long long sC,
                  long long sC2, const float* bias, const float* resid,
                  long long ldr, long long sR, int relu, int outHalf,
                  int storeT, int M, int N, int K, int zdiv, int gz) {
    GemmP p;
    p.A = A; p.Bm = Bm; p.C = C; p.bias = bias; p.resid = resid;
    p.lda = lda; p.ldb = ldb; p.ldc = ldc; p.ldr = ldr;
    p.sA = sA; p.sA2 = sA2; p.sB = sB; p.sB2 = sB2;
    p.sC = sC; p.sC2 = sC2; p.sR = sR;
    p.K = K; p.zdiv = zdiv; p.relu = relu;
    p.outHalf = outHalf; p.storeT = storeT;
    if (narrow) {
      dim3 grid(N / 32, M / 64, gz);
      gemm_wmma<64, 32><<<grid, 64, 0, stream>>>(p);
    } else {
      dim3 grid(N / 64, M / 64, gz);
      gemm_wmma<64, 64><<<grid, 128, 0, stream>>>(p);
    }
  };

  const long long BD = (long long)LDc * Dc;    // per-batch act stride

  // ---- per-node (uu) attention via rank-1 decomposition ----
  gemm(0, UU16, Dc, (long long)LUc * Dc, 0, ciWT, Dc, 0, 0,
       base16, Dc, (long long)LUc * Dc, 0, cib, nullptr, 0, 0, 0, 1, 0,
       LUc, Dc, Dc, 1, Bc);
  gemm(0, DU16, Dc, BD, 0, WT[0], Dc, 0, 0,
       Qb16, Dc, BD, 0, nullptr, nullptr, 0, 0, 0, 1, 0, LDc, Dc, Dc, 1, Bc);
  gemm(0, base16, Dc, (long long)LUc * Dc, 0, WT[1], Dc, 0, 0,
       Kb16, Dc, (long long)LUc * Dc, 0, nullptr, nullptr, 0, 0, 0, 1, 0,
       LUc, Dc, Dc, 1, Bc);
  gemm(0, base16, Dc, (long long)LUc * Dc, 0, WT[2], Dc, 0, 0,
       VbT, LUc, (long long)Dc * LUc, 0, nullptr, nullptr, 0, 0, 0, 1, 1,
       LUc, Dc, Dc, 1, Bc);                                   // V^T [D,LU]
  rowvec_kernel<<<(Dc + 63) / 64, 64, 0, stream>>>(ciW + (long long)Dc * Dc,
                                                   W[1], W[2], wkr, wvr);
  qw_kernel<<<(Bc * LDc * NHc + 255) / 256, 256, 0, stream>>>(Qb16, wkr, qw);
  gemm(0, Qb16, Dc, BD, HDc, Kb16, Dc, (long long)LUc * Dc, HDc,
       S, LUc, (long long)NHc * LDc * LUc, (long long)LDc * LUc,
       nullptr, nullptr, 0, 0, 0, 0, 0, LDc, LUc, HDc, NHc, Bc * NHc);
  softmax_ci_kernel<<<Bc * NHc * LDc, 32, 0, stream>>>(S, CI, qw, att16, sred);
  gemm(1, att16, LUc, (long long)NHc * LDc * LUc, (long long)LDc * LUc,
       VbT, LUc, (long long)Dc * LUc, (long long)HDc * LUc,
       O16, Dc, BD, HDc, nullptr, nullptr, 0, 0, 0, 1, 0,
       LDc, HDc, LUc, NHc, Bc * NHc);
  fixo_kernel<<<(Bc * LDc * Dc + 255) / 256, 256, 0, stream>>>(O16, sred, wvr);
  gemm(0, O16, Dc, BD, 0, WT[3], Dc, 0, 0,
       cat16, 3 * Dc, (long long)LDc * 3 * Dc, 0, nullptr, nullptr, 0, 0,
       0, 1, 0, LDc, Dc, Dc, 1, Bc);

  // ---- ta2du MHA ----
  gemm(0, DU16, Dc, BD, 0, WT[4], Dc, 0, 0,
       Qb16, Dc, BD, 0, nullptr, nullptr, 0, 0, 0, 1, 0, LDc, Dc, Dc, 1, Bc);
  gemm(0, TA16, Dc, (long long)LTc * Dc, 0, WT[5], Dc, 0, 0,
       Kb16, Dc, (long long)LTc * Dc, 0, nullptr, nullptr, 0, 0, 0, 1, 0,
       LTc, Dc, Dc, 1, Bc);
  gemm(0, TA16, Dc, (long long)LTc * Dc, 0, WT[6], Dc, 0, 0,
       VbT, LTc, (long long)Dc * LTc, 0, nullptr, nullptr, 0, 0, 0, 1, 1,
       LTc, Dc, Dc, 1, Bc);
  gemm(0, Qb16, Dc, BD, HDc, Kb16, Dc, (long long)LTc * Dc, HDc,
       S, LTc, (long long)NHc * LDc * LTc, (long long)LDc * LTc,
       nullptr, nullptr, 0, 0, 0, 0, 0, LDc, LTc, HDc, NHc, Bc * NHc);
  softmax_rows_kernel<<<Bc * NHc * LDc, 32, 0, stream>>>(S, att16, LTc, SCALEc);
  gemm(1, att16, LTc, (long long)NHc * LDc * LTc, (long long)LDc * LTc,
       VbT, LTc, (long long)Dc * LTc, (long long)HDc * LTc,
       O16, Dc, BD, HDc, nullptr, nullptr, 0, 0, 0, 1, 0,
       LDc, HDc, LTc, NHc, Bc * NHc);
  gemm(0, O16, Dc, BD, 0, WT[7], Dc, 0, 0,
       cat16 + Dc, 3 * Dc, (long long)LDc * 3 * Dc, 0, nullptr, nullptr, 0, 0,
       0, 1, 0, LDc, Dc, Dc, 1, Bc);

  // ---- du2du MHA ----
  gemm(0, DU16, Dc, BD, 0, WT[8], Dc, 0, 0,
       Qb16, Dc, BD, 0, nullptr, nullptr, 0, 0, 0, 1, 0, LDc, Dc, Dc, 1, Bc);
  gemm(0, DU16, Dc, BD, 0, WT[9], Dc, 0, 0,
       Kb16, Dc, BD, 0, nullptr, nullptr, 0, 0, 0, 1, 0, LDc, Dc, Dc, 1, Bc);
  gemm(0, DU16, Dc, BD, 0, WT[10], Dc, 0, 0,
       VbT, LDc, (long long)Dc * LDc, 0, nullptr, nullptr, 0, 0, 0, 1, 1,
       LDc, Dc, Dc, 1, Bc);
  gemm(0, Qb16, Dc, BD, HDc, Kb16, Dc, BD, HDc,
       S, LDc, (long long)NHc * LDc * LDc, (long long)LDc * LDc,
       nullptr, nullptr, 0, 0, 0, 0, 0, LDc, LDc, HDc, NHc, Bc * NHc);
  softmax_rows_kernel<<<Bc * NHc * LDc, 32, 0, stream>>>(S, att16, LDc, SCALEc);
  gemm(1, att16, LDc, (long long)NHc * LDc * LDc, (long long)LDc * LDc,
       VbT, LDc, (long long)Dc * LDc, (long long)HDc * LDc,
       O16, Dc, BD, HDc, nullptr, nullptr, 0, 0, 0, 1, 0,
       LDc, HDc, LDc, NHc, Bc * NHc);
  gemm(0, O16, Dc, BD, 0, WT[11], Dc, 0, 0,
       cat16 + 2 * Dc, 3 * Dc, (long long)LDc * 3 * Dc, 0, nullptr, nullptr, 0,
       0, 0, 1, 0, LDc, Dc, Dc, 1, Bc);

  // ---- output head ----
  gemm(0, cat16, 3 * Dc, (long long)LDc * 3 * Dc, 0, woWT, 3 * Dc, 0, 0,
       cato16, Dc, BD, 0, wob, nullptr, 0, 0, 0, 1, 0,
       LDc, Dc, 3 * Dc, 1, Bc);
  gemm(0, cato16, Dc, BD, 0, rnW1T, Dc, 0, 0,
       h116, Hc, (long long)LDc * Hc, 0, rnb1, nullptr, 0, 0, 1, 1, 0,
       LDc, Hc, Dc, 1, Bc);
  gemm(0, h116, Hc, (long long)LDc * Hc, 0, rnW2T, Hc, 0, 0,
       rn16, Dc, BD, 0, rnb2, DU, Dc, BD, 0, 1, 0,
       LDc, Dc, Hc, 1, Bc);
  gemm(0, rn16, Dc, BD, 0, poWT, Dc, 0, 0,
       out, Dc, BD, 0, pob, nullptr, 0, 0, 0, 0, 0,
       LDc, Dc, Dc, 1, Bc);
}
